// attention_model_120259084744
// MI455X (gfx1250) — matrix operand order repaired and checked by reading, not compiled, not run on hardware
//
#include <hip/hip_runtime.h>
#include <stdint.h>
#include <stddef.h>

#define B_  64
#define TX_ 512
#define D_  64
#define H_  1024
#define U_  512
#define TY_ 16
#define FH_ 4096   // 4*H

typedef unsigned short u16;
typedef unsigned int   u32;
typedef __attribute__((ext_vector_type(8)))  u16    u16x8;
typedef __attribute__((ext_vector_type(16))) u16    u16x16;
typedef __attribute__((ext_vector_type(16))) __bf16 v16bf;
typedef __attribute__((ext_vector_type(8)))  float  v8f;
typedef __attribute__((ext_vector_type(4)))  u32    u32x4;
typedef __attribute__((ext_vector_type(8)))  u32    u32x8;

static __device__ __forceinline__ u16 f2bf(float f) {
  unsigned u = __builtin_bit_cast(unsigned, f);
  u += 0x7fffu + ((u >> 16) & 1u);            // round-to-nearest-even
  return (u16)(u >> 16);
}
static __device__ __forceinline__ float bf2f(u16 v) {
  return __builtin_bit_cast(float, (unsigned)v << 16);
}
static __device__ __forceinline__ float sigmoidf_(float x) {
  return 1.0f / (1.0f + expf(-x));
}

// ---- WMMA fragment loaders (wave32, 16x16x32 bf16) -------------------------
// A: 16x32 bf16 from row-major source. Lane L<16: row=m0+L, K = k0+[0..7],[16..23]
//                                       Lane L>=16: row=m0+L-16, K = k0+[8..15],[24..31]
static __device__ __forceinline__ v16bf load_a(const u16* base, int rowStride,
                                               int m0, int k0, int lane) {
  int r  = m0 + (lane & 15);
  int kb = k0 + ((lane >> 4) << 3);
  const u16* p = base + (size_t)r * rowStride + kb;
  u16x8 lo = *(const u16x8*)p;
  u16x8 hi = *(const u16x8*)(p + 16);
  u16x16 cat = __builtin_shufflevector(lo, hi,
      0,1,2,3,4,5,6,7,8,9,10,11,12,13,14,15);
  return __builtin_bit_cast(v16bf, cat);
}
// B: 32x16 bf16, sourced from pre-transposed weights WT (N rows x K cols, row-major).
// Lane L<16: col=n0+L, K=k0+[0..7],[16..23]; Lane L>=16: col=n0+L-16, K=k0+[8..15],[24..31]
static __device__ __forceinline__ v16bf load_b(const u16* wt, int ldk,
                                               int n0, int k0, int lane) {
  int col = n0 + (lane & 15);
  int kb  = k0 + ((lane >> 4) << 3);
  const u16* p = wt + (size_t)col * ldk + kb;
  u16x8 lo = *(const u16x8*)p;
  u16x8 hi = *(const u16x8*)(p + 16);
  u16x16 raw = __builtin_shufflevector(lo, hi,
      0,1,2,3,4,5,6,7,8,9,10,11,12,13,14,15);
  return __builtin_bit_cast(v16bf, raw);
}
static __device__ __forceinline__ v8f wmma_bf16(v16bf a, v16bf b, v8f c) {
  return __builtin_amdgcn_wmma_f32_16x16x32_bf16(false, a, false, b,
                                                 (short)0, c, false, false);
}

// ---- conversion kernels ----------------------------------------------------
__global__ void kCvtBf(const float* __restrict__ in, u16* __restrict__ out, int n) {
  int i = blockIdx.x * 256 + threadIdx.x;
  if (i < n) out[i] = f2bf(in[i]);
}
// WT[n*K + k] = bf16( W[(k+rowOff)*N + n] ),  W is (K,N) row-major
__global__ void kCvtT(const float* __restrict__ W, u16* __restrict__ WT,
                      int K, int N, int rowOff) {
  int i = blockIdx.x * 256 + threadIdx.x;
  if (i >= N * K) return;
  int n = i / K, k = i - n * K;
  WT[i] = f2bf(W[(size_t)(k + rowOff) * N + n]);
}
__global__ void kInitState(u16* hA, u16* hB, float* c, int n) {
  int i = blockIdx.x * 256 + threadIdx.x;
  if (i < n) { hA[i] = 0; hB[i] = 0; c[i] = 0.0f; }
}
__global__ void kInitY(const float* __restrict__ x, float* __restrict__ yprev) {
  int b = threadIdx.x;
  if (b < B_) yprev[b] = x[((size_t)b * TX_ + (TX_ - 1)) * D_ + (D_ - 1)];
}

// ---- fused encoder LSTM step: z = h@rk + x_t@k + b; gates; h,c update ------
// grid 32 x 256 (8 waves/block) = 256 waves = 4 Mtiles x 64 Ntiles(within H)
__global__ void kEncStep(const u16* __restrict__ x_bf, const u16* __restrict__ hin,
                         u16* __restrict__ hout, float* __restrict__ c,
                         const u16* __restrict__ enc_kT, const u16* __restrict__ enc_rkT,
                         const float* __restrict__ bias, u16* __restrict__ y_enc, int t) {
  int lane = threadIdx.x & 31;
  int wid  = blockIdx.x * 8 + (threadIdx.x >> 5);
  int mt = wid >> 6, nt = wid & 63;
  int m0 = mt * 16, n0 = nt * 16;
  v8f acc[4];
  for (int g = 0; g < 4; ++g)
    for (int v = 0; v < 8; ++v) acc[g][v] = 0.0f;

#pragma unroll 2
  for (int kc = 0; kc < H_ / 32; ++kc) {          // recurrent term, K=1024
    int k0 = kc * 32;
    v16bf a = load_a(hin, H_, m0, k0, lane);
    for (int g = 0; g < 4; ++g)
      acc[g] = wmma_bf16(a, load_b(enc_rkT, H_, g * H_ + n0, k0, lane), acc[g]);
  }
#pragma unroll
  for (int kc = 0; kc < D_ / 32; ++kc) {          // input term, K=64
    int k0 = kc * 32;
    v16bf a = load_a(x_bf + (size_t)t * D_, TX_ * D_, m0, k0, lane);
    for (int g = 0; g < 4; ++g)
      acc[g] = wmma_bf16(a, load_b(enc_kT, D_, g * H_ + n0, k0, lane), acc[g]);
  }

  int col   = n0 + (lane & 15);
  int rbase = m0 + ((lane >> 4) << 3);
  for (int v = 0; v < 8; ++v) {
    int row = rbase + v;
    float zi = acc[0][v] + bias[0 * H_ + col];
    float zf = acc[1][v] + bias[1 * H_ + col];
    float zg = acc[2][v] + bias[2 * H_ + col];
    float zo = acc[3][v] + bias[3 * H_ + col];
    int idx = row * H_ + col;
    float c2 = sigmoidf_(zf) * c[idx] + sigmoidf_(zi) * tanhf(zg);
    float h2 = sigmoidf_(zo) * tanhf(c2);
    c[idx] = c2;
    u16 hb = f2bf(h2);
    hout[idx] = hb;
    y_enc[((size_t)row * TX_ + t) * H_ + col] = hb;
  }
}

// ---- preA = y_enc @ W1a  (M=B*TX=32768, N=U=512, K=H=1024) -----------------
// One block per 16-row M-tile. The Tensor Data Mover DMAs the 16x1024 bf16
// A tile (32KB) into LDS once; all 8 waves then build A fragments from LDS
// and each wave covers 4 N-tiles (8 waves x 64 cols = N=512).
__global__ void kPreA(const u16* __restrict__ y_enc, const u16* __restrict__ att_w1T,
                      float* __restrict__ preA) {
  __shared__ __attribute__((aligned(32))) u16 sA[16 * H_];
  int lane = threadIdx.x & 31;
  int wave = threadIdx.x >> 5;
  int m0 = blockIdx.x * 16;

  if (threadIdx.x < 32) {
    // Build Tensor DMA descriptor (D#) per CDNA5 ISA ch.8: 2D tile of bf16,
    // tile 1024x16 elements, tensor stride 1024, dims 1024 x 32768.
    unsigned lds = (unsigned)(size_t)(void*)sA;
    unsigned long long ga = (unsigned long long)(size_t)(const void*)(y_enc + (size_t)m0 * H_);
    u32x4 g0;
    g0[0] = 1u;                                        // count=1, user mode
    g0[1] = lds;                                       // lds_addr
    g0[2] = (u32)ga;                                   // global_addr[31:0]
    g0[3] = (u32)((ga >> 32) & 0x1FFFFFFu) | 0x80000000u; // addr[56:32] | type=2
    u32x8 g1;
    g1[0] = 0x00010000u;        // workgroup_mask=0, data_size=1 (2 bytes)
    g1[1] = (u32)(H_) << 16;    // tensor_dim0 = 1024 (low 16 -> bits 63:48)
    g1[2] = (u32)(B_ * TX_) << 16; // tensor_dim0 hi=0 | tensor_dim1 lo16 = 32768
    g1[3] = (u32)(H_) << 16;    // tensor_dim1 hi=0 | tile_dim0 = 1024
    g1[4] = 16u;                // tile_dim1 = 16, tile_dim2 = 0
    g1[5] = (u32)H_;            // tensor_dim0_stride low32 = 1024
    g1[6] = 0u;                 // stride hi / tensor_dim1_stride lo
    g1[7] = 0u;                 // tensor_dim1_stride hi
    asm volatile("tensor_load_to_lds %0, %1" :: "s"(g0), "s"(g1) : "memory");
    __builtin_amdgcn_s_wait_tensorcnt(0);
  }
  __syncthreads();

  int nb = wave * 64;
  v8f acc[4];
  for (int j = 0; j < 4; ++j)
    for (int v = 0; v < 8; ++v) acc[j][v] = 0.0f;

  int ar = lane & 15;
  int ak = (lane >> 4) << 3;
#pragma unroll 2
  for (int kc = 0; kc < H_ / 32; ++kc) {
    int k0 = kc * 32;
    const u16* p = &sA[ar * H_ + k0 + ak];
    u16x8 lo = *(const u16x8*)p;
    u16x8 hi = *(const u16x8*)(p + 16);
    u16x16 cat = __builtin_shufflevector(lo, hi,
        0,1,2,3,4,5,6,7,8,9,10,11,12,13,14,15);
    v16bf a = __builtin_bit_cast(v16bf, cat);
    for (int j = 0; j < 4; ++j)
      acc[j] = wmma_bf16(a, load_b(att_w1T, 2 * H_, nb + 16 * j, k0, lane), acc[j]);
  }
  int rbase = m0 + ((lane >> 4) << 3);
  for (int j = 0; j < 4; ++j) {
    int col = nb + 16 * j + (lane & 15);
    for (int v = 0; v < 8; ++v)
      preA[(size_t)(rbase + v) * U_ + col] = acc[j][v];
  }
}

// ---- hterm = h @ W1b + b1  (M=64, N=512, K=1024) ---------------------------
// grid 16 x 256 = 128 waves = 4 Mtiles x 32 Ntiles
__global__ void kHterm(const u16* __restrict__ h_bf, const u16* __restrict__ att_w1T,
                       const float* __restrict__ b1, float* __restrict__ hterm) {
  int lane = threadIdx.x & 31;
  int wid  = blockIdx.x * 8 + (threadIdx.x >> 5);
  int mt = wid >> 5, nt = wid & 31;
  int m0 = mt * 16, n0 = nt * 16;
  v8f acc;
  for (int v = 0; v < 8; ++v) acc[v] = 0.0f;
#pragma unroll 2
  for (int kc = 0; kc < H_ / 32; ++kc) {
    int k0 = kc * 32;
    acc = wmma_bf16(load_a(h_bf, H_, m0, k0, lane),
                    load_b(att_w1T + H_, 2 * H_, n0, k0, lane), acc);
  }
  int col = n0 + (lane & 15);
  int rbase = m0 + ((lane >> 4) << 3);
  for (int v = 0; v < 8; ++v)
    hterm[(rbase + v) * U_ + col] = acc[v] + b1[col];
}

// ---- scores[b,t] = relu( sum_u tanh(preA + hterm) * w2 + b2 ) --------------
__global__ void kScore(const float* __restrict__ preA, const float* __restrict__ hterm,
                       const float* __restrict__ w2, const float* __restrict__ b2,
                       float* __restrict__ scores) {
  int lane = threadIdx.x & 31;
  int w = blockIdx.x * 8 + (threadIdx.x >> 5);
  int b = w >> 9, t = w & (TX_ - 1);
  const float* pa = preA + ((size_t)b * TX_ + t) * U_;
  const float* ht = hterm + b * U_;
  float acc = 0.0f;
#pragma unroll 4
  for (int j = 0; j < U_ / 32; ++j) {
    int u = lane + j * 32;
    acc += tanhf(pa[u] + ht[u]) * w2[u];
  }
  for (int o = 16; o; o >>= 1) acc += __shfl_down(acc, o, 32);
  if (lane == 0) scores[b * TX_ + t] = fmaxf(acc + b2[0], 0.0f);
}

// ---- softmax over TX per batch row; optionally also write final output -----
__global__ void kSoftmax(const float* __restrict__ scores, float* __restrict__ wbuf,
                         float* __restrict__ wout) {
  __shared__ float red[256];
  int b = blockIdx.x, tid = threadIdx.x;
  float s0 = scores[b * TX_ + tid];
  float s1 = scores[b * TX_ + 256 + tid];
  red[tid] = fmaxf(s0, s1);
  __syncthreads();
  for (int o = 128; o; o >>= 1) {
    if (tid < o) red[tid] = fmaxf(red[tid], red[tid + o]);
    __syncthreads();
  }
  float mx = red[0];
  __syncthreads();
  float e0 = expf(s0 - mx), e1 = expf(s1 - mx);
  red[tid] = e0 + e1;
  __syncthreads();
  for (int o = 128; o; o >>= 1) {
    if (tid < o) red[tid] += red[tid + o];
    __syncthreads();
  }
  float inv = 1.0f / red[0];
  float w0 = e0 * inv, w1 = e1 * inv;
  wbuf[b * TX_ + tid] = w0;
  wbuf[b * TX_ + 256 + tid] = w1;
  if (wout) {
    wout[b * TX_ + tid] = w0;
    wout[b * TX_ + 256 + tid] = w1;
  }
}

// ---- ctx[b,h] = sum_t w[b,t] * y_enc[b,t,h] --------------------------------
__global__ void kCtx(const float* __restrict__ wbuf, const u16* __restrict__ y_enc,
                     u16* __restrict__ ctx_bf) {
  int idx = blockIdx.x * 256 + threadIdx.x;
  int b = idx >> 10, h = idx & (H_ - 1);
  const float* wv = wbuf + b * TX_;
  const u16* ye = y_enc + (size_t)b * TX_ * H_ + h;
  float acc = 0.0f;
#pragma unroll 4
  for (int t = 0; t < TX_; ++t) acc += wv[t] * bf2f(ye[(size_t)t * H_]);
  ctx_bf[b * H_ + h] = f2bf(acc);
}

// ---- fused decoder LSTM step ----------------------------------------------
// z = ctx@dec_k[1:] + h@dec_rk + y_prev*dec_k[0] + b ; gates ; h,c update
__global__ void kDecStep(const u16* __restrict__ ctx_bf, const u16* __restrict__ hin,
                         u16* __restrict__ hout, float* __restrict__ c,
                         const u16* __restrict__ dec_kT, const u16* __restrict__ dec_rkT,
                         const float* __restrict__ bias, const float* __restrict__ dk0,
                         const float* __restrict__ yprev) {
  int lane = threadIdx.x & 31;
  int wid  = blockIdx.x * 8 + (threadIdx.x >> 5);
  int mt = wid >> 6, nt = wid & 63;
  int m0 = mt * 16, n0 = nt * 16;
  v8f acc[4];
  for (int g = 0; g < 4; ++g)
    for (int v = 0; v < 8; ++v) acc[g][v] = 0.0f;
#pragma unroll 2
  for (int kc = 0; kc < H_ / 32; ++kc) {          // ctx term
    int k0 = kc * 32;
    v16bf a = load_a(ctx_bf, H_, m0, k0, lane);
    for (int g = 0; g < 4; ++g)
      acc[g] = wmma_bf16(a, load_b(dec_kT, H_, g * H_ + n0, k0, lane), acc[g]);
  }
#pragma unroll 2
  for (int kc = 0; kc < H_ / 32; ++kc) {          // recurrent term
    int k0 = kc * 32;
    v16bf a = load_a(hin, H_, m0, k0, lane);
    for (int g = 0; g < 4; ++g)
      acc[g] = wmma_bf16(a, load_b(dec_rkT, H_, g * H_ + n0, k0, lane), acc[g]);
  }
  int col   = n0 + (lane & 15);
  int rbase = m0 + ((lane >> 4) << 3);
  for (int v = 0; v < 8; ++v) {
    int row = rbase + v;
    float yp = yprev[row];
    float zi = acc[0][v] + bias[0 * H_ + col] + yp * dk0[0 * H_ + col];
    float zf = acc[1][v] + bias[1 * H_ + col] + yp * dk0[1 * H_ + col];
    float zg = acc[2][v] + bias[2 * H_ + col] + yp * dk0[2 * H_ + col];
    float zo = acc[3][v] + bias[3 * H_ + col] + yp * dk0[3 * H_ + col];
    int idx = row * H_ + col;
    float c2 = sigmoidf_(zf) * c[idx] + sigmoidf_(zi) * tanhf(zg);
    float h2 = sigmoidf_(zo) * tanhf(c2);
    c[idx] = c2;
    hout[idx] = f2bf(h2);
  }
}

// ---- y[b] = h[b]·fin_w + fin_b ; update y_prev; optionally write output ----
__global__ void kFinal(const u16* __restrict__ h_bf, const float* __restrict__ fw,
                       const float* __restrict__ fb, float* __restrict__ yprev,
                       float* __restrict__ yout) {
  int b = blockIdx.x, lane = threadIdx.x;
  float acc = 0.0f;
#pragma unroll 4
  for (int j = lane; j < H_; j += 32) acc += bf2f(h_bf[b * H_ + j]) * fw[j];
  for (int o = 16; o; o >>= 1) acc += __shfl_down(acc, o, 32);
  if (lane == 0) {
    float y = acc + fb[0];
    yprev[b] = y;
    if (yout) yout[b] = y;
  }
}

// ---------------------------------------------------------------------------
extern "C" void kernel_launch(void* const* d_in, const int* in_sizes, int n_in,
                              void* d_out, int out_size, void* d_ws, size_t ws_size,
                              hipStream_t stream) {
  (void)in_sizes; (void)n_in; (void)out_size; (void)ws_size;
  const float* input_seq = (const float*)d_in[0];
  const float* enc_k  = (const float*)d_in[1];
  const float* enc_rk = (const float*)d_in[2];
  const float* enc_b  = (const float*)d_in[3];
  const float* dec_k  = (const float*)d_in[4];
  const float* dec_rk = (const float*)d_in[5];
  const float* dec_b  = (const float*)d_in[6];
  const float* att_w1 = (const float*)d_in[7];
  const float* att_b1 = (const float*)d_in[8];
  const float* att_w2 = (const float*)d_in[9];
  const float* att_b2 = (const float*)d_in[10];
  const float* fin_w  = (const float*)d_in[11];
  const float* fin_b  = (const float*)d_in[12];
  float* out = (float*)d_out;

  char* ws = (char*)d_ws;
  size_t off = 0;
  auto alloc = [&](size_t bytes) -> char* {
    char* p = ws + off;
    off = (off + bytes + 255) & ~(size_t)255;
    return p;
  };
  u16*   x_bf    = (u16*)  alloc((size_t)B_ * TX_ * D_ * 2);
  u16*   enc_kT  = (u16*)  alloc((size_t)FH_ * D_ * 2);
  u16*   enc_rkT = (u16*)  alloc((size_t)FH_ * H_ * 2);
  u16*   dec_kT  = (u16*)  alloc((size_t)FH_ * H_ * 2);
  u16*   dec_rkT = (u16*)  alloc((size_t)FH_ * H_ * 2);
  u16*   att_w1T = (u16*)  alloc((size_t)U_ * 2 * H_ * 2);
  u16*   y_enc   = (u16*)  alloc((size_t)B_ * TX_ * H_ * 2);
  float* preA    = (float*)alloc((size_t)B_ * TX_ * U_ * 4);
  u16*   hA      = (u16*)  alloc((size_t)B_ * H_ * 2);
  u16*   hB      = (u16*)  alloc((size_t)B_ * H_ * 2);
  float* cbuf    = (float*)alloc((size_t)B_ * H_ * 4);
  float* hterm   = (float*)alloc((size_t)B_ * U_ * 4);
  float* scores  = (float*)alloc((size_t)B_ * TX_ * 4);
  float* wbuf    = (float*)alloc((size_t)B_ * TX_ * 4);
  u16*   ctx_bf  = (u16*)  alloc((size_t)B_ * H_ * 2);
  float* yprev   = (float*)alloc((size_t)B_ * 4);

  dim3 blk(256);
  kCvtBf<<<(B_ * TX_ * D_ + 255) / 256, blk, 0, stream>>>(input_seq, x_bf, B_ * TX_ * D_);
  kCvtT<<<(FH_ * D_ + 255) / 256, blk, 0, stream>>>(enc_k,  enc_kT,  D_,     FH_, 0);
  kCvtT<<<(FH_ * H_ + 255) / 256, blk, 0, stream>>>(enc_rk, enc_rkT, H_,     FH_, 0);
  kCvtT<<<(FH_ * H_ + 255) / 256, blk, 0, stream>>>(dec_k,  dec_kT,  H_,     FH_, 1);
  kCvtT<<<(FH_ * H_ + 255) / 256, blk, 0, stream>>>(dec_rk, dec_rkT, H_,     FH_, 0);
  kCvtT<<<(U_ * 2 * H_ + 255) / 256, blk, 0, stream>>>(att_w1, att_w1T, 2 * H_, U_, 0);
  kInitState<<<(B_ * H_ + 255) / 256, blk, 0, stream>>>(hA, hB, cbuf, B_ * H_);
  kInitY<<<1, 64, 0, stream>>>(input_seq, yprev);

  u16* hbuf[2] = {hA, hB};
  for (int t = 0; t < TX_; ++t)
    kEncStep<<<32, blk, 0, stream>>>(x_bf, hbuf[t & 1], hbuf[1 - (t & 1)], cbuf,
                                     enc_kT, enc_rkT, enc_b, y_enc, t);
  kPreA<<<(B_ * TX_) / 16, blk, 0, stream>>>(y_enc, att_w1T, preA);

  for (int s = 0; s < TY_; ++s) {
    const u16* hin = hbuf[s & 1];
    u16* hout = hbuf[1 - (s & 1)];
    kHterm<<<16, blk, 0, stream>>>(hin, att_w1T, att_b1, hterm);
    kScore<<<B_ * TX_ / 8, blk, 0, stream>>>(preA, hterm, att_w2, att_b2, scores);
    kSoftmax<<<B_, blk, 0, stream>>>(scores, wbuf, (s == TY_ - 1) ? (out + B_) : nullptr);
    kCtx<<<B_ * H_ / 256, blk, 0, stream>>>(wbuf, y_enc, ctx_bf);
    kDecStep<<<32, blk, 0, stream>>>(ctx_bf, hin, hout, cbuf, dec_kT, dec_rkT,
                                     dec_b, dec_k, yprev);
    kFinal<<<B_, 32, 0, stream>>>(hout, fin_w, fin_b, yprev,
                                  (s == TY_ - 1) ? out : nullptr);
  }
}
